// ShortRec_88046829568499
// MI455X (gfx1250) — matrix operand order repaired and checked by reading, not compiled, not run on hardware
//
#include <hip/hip_runtime.h>
#include <hip/hip_bf16.h>
#include <math.h>

typedef __attribute__((ext_vector_type(16))) _Float16 v16h;
typedef __attribute__((ext_vector_type(8)))  float    v8f;
typedef __attribute__((ext_vector_type(4)))  int      v4i;
typedef __attribute__((ext_vector_type(8)))  int      v8i;
typedef __attribute__((ext_vector_type(4)))  unsigned v4u;

static constexpr int B_  = 512;
static constexpr int S_  = 200;
static constexpr int SP  = 224;     // S padded to 7*32 for WMMA K-chunks
static constexpr int D_  = 128;
static constexpr int NC  = 100000;
static constexpr float LN_EPS = 1e-6f;

// ---------------------------------------------------------------------------
// CDNA5 feature probes (safe: fall back to sync paths if builtins missing)
// ---------------------------------------------------------------------------
#if __has_builtin(__builtin_amdgcn_global_load_async_to_lds_b128) && \
    __has_builtin(__builtin_amdgcn_s_wait_asynccnt)
#define USE_ASYNC 1
#endif

#if __has_builtin(__builtin_amdgcn_tensor_load_to_lds) && \
    __has_builtin(__builtin_amdgcn_s_wait_tensorcnt)
#define USE_TDM 1
#endif

#if __has_builtin(__builtin_amdgcn_tanhf)
#define fast_tanh(x) __builtin_amdgcn_tanhf(x)
#elif __has_builtin(__builtin_amdgcn_tanh_f32)
#define fast_tanh(x) __builtin_amdgcn_tanh_f32(x)
#else
#define fast_tanh(x) tanhf(x)
#endif

// ---------------------------------------------------------------------------
// WMMA helpers (CDNA5 wave32, V_WMMA_F32_16X16X32_F16)
// ---------------------------------------------------------------------------
__device__ __forceinline__ v8f zero8() {
  v8f z;
#pragma unroll
  for (int i = 0; i < 8; ++i) z[i] = 0.f;
  return z;
}

__device__ __forceinline__ v8f wmma16(v16h a, v16h b, v8f c) {
  return __builtin_amdgcn_wmma_f32_16x16x32_f16(false, a, false, b, (short)0, c,
                                                false, false);
}

// A-matrix fragment, 16x32 f16, row-major source, leading dim `ld` halves.
__device__ __forceinline__ v16h load_a(const _Float16* src, int ld, int lane) {
  int m = lane & 15, grp = lane >> 4;
  const _Float16* row = src + m * ld;
  v16h f;
#pragma unroll
  for (int v = 0; v < 8; ++v) {
    int k = grp * 16 + 2 * v;
    union { unsigned u; _Float16 h[2]; } t;
    t.u = *(const unsigned*)(row + k);
    f[2 * v] = t.h[0];
    f[2 * v + 1] = t.h[1];
  }
  return f;
}

// B-matrix fragment (32x16, KxN) from N-major ("transposed") source.
__device__ __forceinline__ v16h load_bt(const _Float16* src, int ld, int lane) {
  int n = lane & 15, grp = lane >> 4;
  const _Float16* row = src + n * ld;
  v16h f;
#pragma unroll
  for (int v = 0; v < 8; ++v) {
    int k = grp * 16 + 2 * v;
    union { unsigned u; _Float16 h[2]; } t;
    t.u = *(const unsigned*)(row + k);
    f[2 * v] = t.h[0];
    f[2 * v + 1] = t.h[1];
  }
  return f;
}

// B-matrix fragment from a K-major (row-major KxN) source.
__device__ __forceinline__ v16h load_bk(const _Float16* src, int ld, int n0,
                                        int lane) {
  int n = n0 + (lane & 15), grp = lane >> 4;
  v16h f;
#pragma unroll
  for (int v = 0; v < 8; ++v) {
    int k = grp * 16 + 2 * v;
    f[2 * v]     = src[k * ld + n];
    f[2 * v + 1] = src[(k + 1) * ld + n];
  }
  return f;
}

// Stage a raw f16 tile global -> LDS.  Async (ASYNCcnt) when available.
__device__ __forceinline__ void stage_tile(_Float16* dst, const _Float16* src,
                                           int halves, int t, int nthreads) {
#ifdef USE_ASYNC
  int vecs = halves >> 3;  // 16-byte units
  for (int i = t; i < vecs; i += nthreads) {
    auto g = (__attribute__((address_space(1))) v4i*)(src + i * 8);
    auto l = (__attribute__((address_space(3))) v4i*)(dst + i * 8);
    __builtin_amdgcn_global_load_async_to_lds_b128(g, l, 0, 0);
  }
  __builtin_amdgcn_s_wait_asynccnt(0);
#else
  const unsigned* s = (const unsigned*)src;
  unsigned* d = (unsigned*)dst;
  for (int i = t; i < (halves >> 1); i += nthreads) d[i] = s[i];
  asm volatile("s_wait_dscnt 0" ::: "memory");
#endif
}

#ifdef USE_TDM
// One-shot 1-D TDM load of `nelem` f16 elements global -> LDS (wave-level,
// tracked by TENSORcnt).  D# built per CDNA5 ISA 8.3/8.4; 6-arg builtin
// (uint32x4 g0, int32x8 g1, int32x4 g2, int32x4 g3, int32x8, i32 cpol).
__device__ __forceinline__ void tdm_load_f16(_Float16* dst,
                                             const _Float16* src,
                                             unsigned nelem) {
  unsigned laddr =
      (unsigned)(size_t)(__attribute__((address_space(3))) _Float16*)dst;
  unsigned long long ga = (unsigned long long)(size_t)src;
  v4u g0;
  g0[0] = 1u;                                   // count=1 (valid), user mode
  g0[1] = laddr;                                // lds_addr
  g0[2] = (unsigned)(ga & 0xffffffffu);         // global_addr[31:0]
  g0[3] = (unsigned)((ga >> 32) & 0x1ffffffu)   // global_addr[56:32]
          | (2u << 30);                         // type = 2 ("image")
  v8i g1;
  g1[0] = 0x00010000;                           // data_size=1 -> 2 bytes
  g1[1] = (int)((nelem & 0xffffu) << 16);       // tensor_dim0[15:0]  @[63:48]
  g1[2] = (int)((nelem >> 16) | (1u << 16));    // tdim0[31:16], tensor_dim1=1
  g1[3] = (int)((nelem & 0xffffu) << 16);       // tile_dim0          @[127:112]
  g1[4] = 0;                                    // tile_dim1/2 unused
  g1[5] = (int)nelem;                           // tensor_dim0_stride[31:0]
  g1[6] = 0;
  g1[7] = 0;
  v4i zz = {0, 0, 0, 0};
  v8i z8 = {0, 0, 0, 0, 0, 0, 0, 0};
  __builtin_amdgcn_tensor_load_to_lds(g0, g1, zz, zz, z8, 0);
}
#endif

// Order-preserving float <-> uint key (atomic row max of signed floats)
__device__ __forceinline__ unsigned fkey(float f) {
  unsigned u = __float_as_uint(f);
  return (u & 0x80000000u) ? ~u : (u | 0x80000000u);
}
__device__ __forceinline__ float funkey(unsigned k) {
  unsigned u = (k & 0x80000000u) ? (k & 0x7fffffffu) : ~k;
  return __uint_as_float(u);
}

// ---------------------------------------------------------------------------
// 0) init accumulators (must re-init every call; no cross-call state)
// ---------------------------------------------------------------------------
__global__ __launch_bounds__(256) void init_kernel(float* pooled,
                                                   unsigned* rowmax_u,
                                                   float* rowsum) {
  int i = blockIdx.x * 256 + threadIdx.x;
  if (i < B_ * D_) pooled[i] = 0.f;
  if (i < B_) { rowmax_u[i] = 0u; rowsum[i] = 0.f; }
}

// ---------------------------------------------------------------------------
// 1) prep: f16 N-major (transposed) weight copies for WMMA B fragments
// ---------------------------------------------------------------------------
__global__ __launch_bounds__(256) void prep_kernel(
    const float* __restrict__ Wq, const float* __restrict__ Wk,
    const float* __restrict__ Wv, const float* __restrict__ Wf,
    const float* __restrict__ rec, _Float16* WqT, _Float16* WkT, _Float16* WvT,
    _Float16* WfT, _Float16* recT) {
  int i = blockIdx.x * 256 + threadIdx.x;
  if (i < D_ * D_) {
    int n = i >> 7, k = i & 127;
    WqT[i] = (_Float16)Wq[k * D_ + n];
    WkT[i] = (_Float16)Wk[k * D_ + n];
    WvT[i] = (_Float16)Wv[k * D_ + n];
    WfT[i] = (_Float16)Wf[k * D_ + n];
  }
  if (i < 3 * D_ * D_) {           // recT[n][k] = rec[k][n]
    int n = i >> 7, k = i & 127;
    recT[i] = (_Float16)rec[k * 384 + n];
  }
}

// ---------------------------------------------------------------------------
// 2) GRU (Keras reset_after): sequential over S, parallel over B.
//    32 blocks x 16 batch rows, 8 waves split the 24 N-tiles of h@Wrec.
//    Serial-path: h16 updated inside the gate loop (one barrier saved/step),
//    time_seq preloaded to LDS, v_rcp_f32 / v_tanh_f32 gates.
// ---------------------------------------------------------------------------
__global__ __launch_bounds__(256) void gru_kernel(
    const float* __restrict__ time_seq, const float* __restrict__ gk,
    const float* __restrict__ grub, const _Float16* __restrict__ recT,
    float* __restrict__ time_emb) {
  __shared__ alignas(16) float    hbuf[16 * D_];
  __shared__ alignas(16) _Float16 h16[16 * D_];
  __shared__ alignas(16) float    mh[16 * 384];
  __shared__ alignas(16) float    ts_s[16 * S_];

  int tid = threadIdx.x, lane = tid & 31, wv = tid >> 5;
  int b0 = blockIdx.x * 16;
  const float* bi = grub;          // input bias  [384]
  const float* br = grub + 384;    // recurrent bias [384]

  for (int i = tid; i < 16 * D_; i += 256) {
    hbuf[i] = 0.f;
    h16[i] = (_Float16)0.f;
  }
  for (int i = tid; i < 16 * S_; i += 256)
    ts_s[i] = time_seq[(b0 + i / S_) * S_ + (i % S_)];
  __syncthreads();

  for (int s = 0; s < S_; ++s) {
    v16h a[4];
#pragma unroll
    for (int kc = 0; kc < 4; ++kc) a[kc] = load_a(h16 + kc * 32, D_, lane);

    for (int t = wv * 3; t < wv * 3 + 3; ++t) {   // this wave's 3 N-tiles
      v8f acc = zero8();
#pragma unroll
      for (int kc = 0; kc < 4; ++kc)
        acc = wmma16(a[kc], load_bt(recT + t * 16 * D_ + kc * 32, D_, lane), acc);
      int n = lane & 15, grp = lane >> 4;
#pragma unroll
      for (int r = 0; r < 8; ++r)
        mh[(r + grp * 8) * 384 + t * 16 + n] = acc[r];
    }
    __syncthreads();

    for (int e = tid; e < 16 * D_; e += 256) {
      int row = e >> 7, d = e & 127;
      float xt = ts_s[row * S_ + s];
      float xz = xt * gk[d]       + bi[d];
      float xr = xt * gk[128 + d] + bi[128 + d];
      float xh = xt * gk[256 + d] + bi[256 + d];
      float hz = mh[row * 384 + d]       + br[d];
      float hr = mh[row * 384 + 128 + d] + br[128 + d];
      float hh = mh[row * 384 + 256 + d] + br[256 + d];
      float z = __builtin_amdgcn_rcpf(1.f + __expf(-(xz + hz)));
      float r = __builtin_amdgcn_rcpf(1.f + __expf(-(xr + hr)));
      float c = fast_tanh(xh + r * hh);
      float hn = z * hbuf[e] + (1.f - z) * c;
      hbuf[e] = hn;
      h16[e] = (_Float16)hn;
      time_emb[((size_t)(b0 + row) * S_ + s) * D_ + d] = hn;
    }
    __syncthreads();
  }
}

// ---------------------------------------------------------------------------
// 3) QKV projections (+ K/V additive time & freq embeddings), f16 output,
//    S padded to 224 with zeros.  One wave per (b, 16-token tile).
// ---------------------------------------------------------------------------
__global__ __launch_bounds__(32) void qkv_kernel(
    const int* __restrict__ item_seq, const int* __restrict__ freq_seq,
    const float* __restrict__ emb_item, const float* __restrict__ emb_freq,
    const float* __restrict__ time_emb, const _Float16* __restrict__ WqT,
    const _Float16* __restrict__ WkT, const _Float16* __restrict__ WvT,
    const float* __restrict__ bq, const float* __restrict__ bk,
    const float* __restrict__ bv, _Float16* __restrict__ Qh,
    _Float16* __restrict__ Kh, _Float16* __restrict__ Vh) {
  __shared__ alignas(16) _Float16 xi[16 * D_];
  int lane = threadIdx.x;
  int st = blockIdx.x % 14, b = blockIdx.x / 14, s0 = st * 16;

  for (int e = lane; e < 16 * D_; e += 32) {
    int row = e >> 7, d = e & 127, s = s0 + row;
    float v = 0.f;
    if (s < S_) {
      int idx = item_seq[b * S_ + s];
      v = emb_item[(size_t)idx * D_ + d];
    }
    xi[e] = (_Float16)v;
  }
  __syncthreads();

  v16h a[4];
#pragma unroll
  for (int kc = 0; kc < 4; ++kc) a[kc] = load_a(xi + kc * 32, D_, lane);
  int n = lane & 15, grp = lane >> 4;

  for (int nt = 0; nt < 8; ++nt) {
    v8f aq = zero8(), ak = zero8(), av = zero8();
#pragma unroll
    for (int kc = 0; kc < 4; ++kc) {
      aq = wmma16(a[kc], load_bt(WqT + nt * 16 * D_ + kc * 32, D_, lane), aq);
      ak = wmma16(a[kc], load_bt(WkT + nt * 16 * D_ + kc * 32, D_, lane), ak);
      av = wmma16(a[kc], load_bt(WvT + nt * 16 * D_ + kc * 32, D_, lane), av);
    }
    int col = nt * 16 + n;
#pragma unroll
    for (int r = 0; r < 8; ++r) {
      int m = r + grp * 8, s = s0 + m;
      size_t po = ((size_t)b * SP + s0 + m) * D_ + col;
      if (s < S_) {
        size_t so = ((size_t)b * S_ + s) * D_ + col;
        float te = time_emb[so];
        int fidx = freq_seq[b * S_ + s];
        float fe = emb_freq[(size_t)fidx * D_ + col];
        Qh[po] = (_Float16)(aq[r] + bq[col]);
        Kh[po] = (_Float16)(ak[r] + bk[col] + te + fe);
        Vh[po] = (_Float16)(av[r] + bv[col] + te + fe);
      } else {
        Qh[po] = (_Float16)0.f;
        Kh[po] = (_Float16)0.f;
        Vh[po] = (_Float16)0.f;
      }
    }
  }
}

// ---------------------------------------------------------------------------
// 4) Fused attention + FFN + LayerNorm + mean-pool.  128 threads = 4 waves.
//    K tiles: TDM tensor_load_to_lds (TENSORcnt).  Q/V: async-to-LDS.
// ---------------------------------------------------------------------------
__global__ __launch_bounds__(128) void attn_kernel(
    const _Float16* __restrict__ Qh, const _Float16* __restrict__ Kh,
    const _Float16* __restrict__ Vh, const _Float16* __restrict__ WfT,
    const float* __restrict__ bf, const float* __restrict__ gamma,
    const float* __restrict__ beta, float* __restrict__ pooled) {
  __shared__ alignas(16) _Float16 qs[16 * D_];     // Q tile / attn_out f16
  __shared__ alignas(16) _Float16 ks[4][16 * D_];  // wave-private K tiles
  __shared__ alignas(16) _Float16 vs[32 * D_];
  __shared__ alignas(16) float    lg[16 * SP];
  __shared__ alignas(16) _Float16 ps[16 * SP];
  __shared__ alignas(16) float    xs[16 * D_];
  __shared__ float mu_s[16], rs_s[16];

  int tid = threadIdx.x, lane = tid & 31, wv = tid >> 5;
  int qt = blockIdx.x % 13, b = blockIdx.x / 13, q0 = qt * 16;
  const size_t base = (size_t)b * SP * D_;
  int n = lane & 15, grp = lane >> 4;
  const float scale = 0.08838834764831845f;  // 1/sqrt(128)

  stage_tile(qs, Qh + base + (size_t)q0 * D_, 16 * D_, tid, 128);
  __syncthreads();
  v16h aq[4];
#pragma unroll
  for (int kc = 0; kc < 4; ++kc) aq[kc] = load_a(qs + kc * 32, D_, lane);

  // ---- logits = Q K^T * scale; wave w owns key tiles {w, w+4, w+8, w+12} ----
  for (int kt = wv; kt < 14; kt += 4) {
#ifdef USE_TDM
    tdm_load_f16(ks[wv], Kh + base + (size_t)kt * 16 * D_, 16 * D_);
    __builtin_amdgcn_s_wait_tensorcnt(0);
#else
    stage_tile(ks[wv], Kh + base + (size_t)kt * 16 * D_, 16 * D_, lane, 32);
#endif
    v8f acc = zero8();
#pragma unroll
    for (int kc = 0; kc < 4; ++kc)
      acc = wmma16(aq[kc], load_bt(ks[wv] + kc * 32, D_, lane), acc);
#pragma unroll
    for (int r = 0; r < 8; ++r) {
      int m = r + grp * 8, key = kt * 16 + n;
      lg[m * SP + key] = (key < S_) ? acc[r] * scale : -1e30f;
    }
  }
  __syncthreads();

  // ---- row softmax -> normalized P (f16); padded keys -> 0 ----
  if (tid < 16) {
    float mx = -1e30f;
    for (int j = 0; j < SP; ++j) mx = fmaxf(mx, lg[tid * SP + j]);
    float sum = 0.f;
    for (int j = 0; j < SP; ++j) sum += __expf(lg[tid * SP + j] - mx);
    float inv = 1.f / sum;
    for (int j = 0; j < SP; ++j)
      ps[tid * SP + j] = (_Float16)(__expf(lg[tid * SP + j] - mx) * inv);
  }
  __syncthreads();

  // ---- attn_out = P @ V  (7 K-chunks; each wave owns 2 of 8 N-tiles) ----
  v8f ao0 = zero8(), ao1 = zero8();
  for (int ch = 0; ch < 7; ++ch) {
    stage_tile(vs, Vh + base + (size_t)ch * 32 * D_, 32 * D_, tid, 128);
    __syncthreads();
    v16h ap = load_a(ps + ch * 32, SP, lane);
    ao0 = wmma16(ap, load_bk(vs, D_, (2 * wv) * 16, lane), ao0);
    ao1 = wmma16(ap, load_bk(vs, D_, (2 * wv + 1) * 16, lane), ao1);
    __syncthreads();
  }
#pragma unroll
  for (int r = 0; r < 8; ++r) {
    xs[(r + grp * 8) * D_ + (2 * wv) * 16 + n]     = ao0[r];
    xs[(r + grp * 8) * D_ + (2 * wv + 1) * 16 + n] = ao1[r];
  }
  __syncthreads();

  // ---- FFN: x = attn + relu(attn @ Wf + bf); each wave owns 2 N-tiles ----
  for (int e = tid; e < 16 * D_; e += 128) qs[e] = (_Float16)xs[e];
  __syncthreads();
  v16h aa[4];
#pragma unroll
  for (int kc = 0; kc < 4; ++kc) aa[kc] = load_a(qs + kc * 32, D_, lane);
#pragma unroll
  for (int j = 0; j < 2; ++j) {
    int nt = 2 * wv + j;
    v8f f = zero8();
#pragma unroll
    for (int kc = 0; kc < 4; ++kc)
      f = wmma16(aa[kc], load_bt(WfT + nt * 16 * D_ + kc * 32, D_, lane), f);
    int col = nt * 16 + n;
#pragma unroll
    for (int r = 0; r < 8; ++r)
      xs[(r + grp * 8) * D_ + col] += fmaxf(f[r] + bf[col], 0.f);
  }
  __syncthreads();

  // ---- LayerNorm stats per row ----
  if (tid < 16) {
    float m_ = 0.f;
    for (int d = 0; d < D_; ++d) m_ += xs[tid * D_ + d];
    m_ *= (1.f / D_);
    float v_ = 0.f;
    for (int d = 0; d < D_; ++d) {
      float t = xs[tid * D_ + d] - m_;
      v_ += t * t;
    }
    v_ *= (1.f / D_);
    mu_s[tid] = m_;
    rs_s[tid] = rsqrtf(v_ + LN_EPS);
  }
  __syncthreads();

  // ---- pool: accumulate mean over valid s into pooled[b]; d == tid ----
  {
    int d = tid;
    float sum = 0.f;
    for (int m = 0; m < 16; ++m) {
      if (q0 + m < S_)
        sum += (xs[m * D_ + d] - mu_s[m]) * rs_s[m] * gamma[d] + beta[d];
    }
    atomicAdd(&pooled[b * D_ + d], sum * (1.f / (float)S_));
  }
}

// ---------------------------------------------------------------------------
// 5) pooled f32 -> f16
// ---------------------------------------------------------------------------
__global__ __launch_bounds__(256) void p16_kernel(const float* __restrict__ p,
                                                  _Float16* __restrict__ p16) {
  int i = blockIdx.x * 256 + threadIdx.x;
  if (i < B_ * D_) p16[i] = (_Float16)p[i];
}

// ---------------------------------------------------------------------------
// 6) classifier: logits = pooled @ Wo + bo; Wo tile read once per block,
//    8 waves cover all 512 batch rows; keyed atomic row-max.
// ---------------------------------------------------------------------------
__global__ __launch_bounds__(256) void cls_kernel(
    const _Float16* __restrict__ pooled16, const float* __restrict__ Wo,
    const float* __restrict__ bo, float* __restrict__ out,
    unsigned* __restrict__ rowmax_u) {
  __shared__ alignas(16) _Float16 wot[16 * D_];
  __shared__ alignas(16) float tile[8][16 * 16];
  int tid = threadIdx.x, lane = tid & 31, wv = tid >> 5;
  int n0 = blockIdx.x * 16;

  // prefetch the next tile's Wo columns while we convert this one
  if (tid < 128 && n0 + 16 < NC)
    __builtin_prefetch(Wo + (size_t)tid * NC + n0 + 16, 0, 0);

  for (int e = tid; e < 16 * D_; e += 256) {
    int nn = e >> 7, k = e & 127;
    wot[e] = (_Float16)Wo[(size_t)k * NC + n0 + nn];
  }
  __syncthreads();

  int n = lane & 15, grp = lane >> 4;
  for (int it = 0; it < 4; ++it) {  // uniform trip count across all 8 waves
    int mt = wv * 4 + it, m0 = mt * 16;
    v8f acc = zero8();
#pragma unroll
    for (int kc = 0; kc < 4; ++kc) {
      v16h a = load_a(pooled16 + (size_t)m0 * D_ + kc * 32, D_, lane);
      acc = wmma16(a, load_bt(wot + kc * 32, D_, lane), acc);
    }
    float bb = bo[n0 + n];
#pragma unroll
    for (int r = 0; r < 8; ++r) {
      int m = m0 + r + grp * 8;
      float v = acc[r] + bb;
      out[(size_t)m * NC + n0 + n] = v;
      tile[wv][(r + grp * 8) * 16 + n] = v;
    }
    __syncthreads();
    if (lane < 16) {
      float mx = -1e30f;
      for (int j = 0; j < 16; ++j) mx = fmaxf(mx, tile[wv][lane * 16 + j]);
      atomicMax(&rowmax_u[m0 + lane], fkey(mx));
    }
    __syncthreads();
  }
}

// ---------------------------------------------------------------------------
// 7) exp(l - rowmax) in place + per-row sum (b128 traffic); 8) normalize
// ---------------------------------------------------------------------------
__global__ __launch_bounds__(256) void expsum_kernel(
    float* __restrict__ out, const unsigned* __restrict__ rowmax_u,
    float* __restrict__ rowsum) {
  __shared__ float red[256];
  int row = blockIdx.y;
  int c0 = blockIdx.x * 2048 + threadIdx.x * 4;
  float mx = funkey(rowmax_u[row]);
  float s = 0.f;
#pragma unroll
  for (int i = 0; i < 2; ++i) {
    int c = c0 + i * 1024;
    if (c < NC) {  // NC % 4 == 0, c % 4 == 0: no partial vectors
      float4* p4 = (float4*)(out + (size_t)row * NC + c);
      float4 v = *p4;
      v.x = __expf(v.x - mx);
      v.y = __expf(v.y - mx);
      v.z = __expf(v.z - mx);
      v.w = __expf(v.w - mx);
      *p4 = v;
      s += v.x + v.y + v.z + v.w;
    }
  }
  red[threadIdx.x] = s;
  __syncthreads();
  for (int st = 128; st > 0; st >>= 1) {
    if (threadIdx.x < st) red[threadIdx.x] += red[threadIdx.x + st];
    __syncthreads();
  }
  if (threadIdx.x == 0) atomicAdd(&rowsum[row], red[0]);
}

__global__ __launch_bounds__(256) void norm_kernel(
    float* __restrict__ out, const float* __restrict__ rowsum) {
  int row = blockIdx.y;
  int c0 = blockIdx.x * 2048 + threadIdx.x * 4;
  float inv = 1.f / rowsum[row];
#pragma unroll
  for (int i = 0; i < 2; ++i) {
    int c = c0 + i * 1024;
    if (c < NC) {
      float4* p4 = (float4*)(out + (size_t)row * NC + c);
      float4 v = *p4;
      v.x *= inv; v.y *= inv; v.z *= inv; v.w *= inv;
      *p4 = v;
    }
  }
}

// ---------------------------------------------------------------------------
extern "C" void kernel_launch(void* const* d_in, const int* in_sizes, int n_in,
                              void* d_out, int out_size, void* d_ws,
                              size_t ws_size, hipStream_t stream) {
  const int*   item_seq = (const int*)d_in[0];
  const float* time_seq = (const float*)d_in[1];
  const int*   freq_seq = (const int*)d_in[2];
  const float* emb_item = (const float*)d_in[3];
  const float* emb_freq = (const float*)d_in[4];
  const float* gru_k    = (const float*)d_in[5];
  const float* gru_rec  = (const float*)d_in[6];
  const float* gru_bias = (const float*)d_in[7];
  const float* Wq = (const float*)d_in[8];  const float* bq = (const float*)d_in[9];
  const float* Wk = (const float*)d_in[10]; const float* bk = (const float*)d_in[11];
  const float* Wv = (const float*)d_in[12]; const float* bv = (const float*)d_in[13];
  const float* Wf = (const float*)d_in[14]; const float* bf = (const float*)d_in[15];
  const float* gamma = (const float*)d_in[16];
  const float* beta  = (const float*)d_in[17];
  const float* Wo = (const float*)d_in[18]; const float* bo = (const float*)d_in[19];
  float* out = (float*)d_out;

  // workspace carve-out (256B aligned)
  char* p = (char*)d_ws;
  auto alloc = [&](size_t bytes) {
    char* r = p;
    p += (bytes + 255) & ~(size_t)255;
    return r;
  };
  float*    time_emb = (float*)alloc((size_t)B_ * S_ * D_ * 4);
  _Float16* Qh   = (_Float16*)alloc((size_t)B_ * SP * D_ * 2);
  _Float16* Kh   = (_Float16*)alloc((size_t)B_ * SP * D_ * 2);
  _Float16* Vh   = (_Float16*)alloc((size_t)B_ * SP * D_ * 2);
  _Float16* WqT  = (_Float16*)alloc((size_t)D_ * D_ * 2);
  _Float16* WkT  = (_Float16*)alloc((size_t)D_ * D_ * 2);
  _Float16* WvT  = (_Float16*)alloc((size_t)D_ * D_ * 2);
  _Float16* WfT  = (_Float16*)alloc((size_t)D_ * D_ * 2);
  _Float16* recT = (_Float16*)alloc((size_t)3 * D_ * D_ * 2);
  float*    pooled   = (float*)alloc((size_t)B_ * D_ * 4);
  _Float16* pooled16 = (_Float16*)alloc((size_t)B_ * D_ * 2);
  unsigned* rowmax_u = (unsigned*)alloc((size_t)B_ * 4);
  float*    rowsum   = (float*)alloc((size_t)B_ * 4);

  init_kernel<<<256, 256, 0, stream>>>(pooled, rowmax_u, rowsum);
  prep_kernel<<<192, 256, 0, stream>>>(Wq, Wk, Wv, Wf, gru_rec, WqT, WkT, WvT,
                                       WfT, recT);
  gru_kernel<<<B_ / 16, 256, 0, stream>>>(time_seq, gru_k, gru_bias, recT,
                                          time_emb);
  qkv_kernel<<<B_ * 14, 32, 0, stream>>>(item_seq, freq_seq, emb_item, emb_freq,
                                         time_emb, WqT, WkT, WvT, bq, bk, bv,
                                         Qh, Kh, Vh);
  attn_kernel<<<B_ * 13, 128, 0, stream>>>(Qh, Kh, Vh, WfT, bf, gamma, beta,
                                           pooled);
  p16_kernel<<<256, 256, 0, stream>>>(pooled, pooled16);
  cls_kernel<<<NC / 16, 256, 0, stream>>>(pooled16, Wo, bo, out, rowmax_u);
  dim3 g((NC + 2047) / 2048, B_);
  expsum_kernel<<<g, 256, 0, stream>>>(out, rowmax_u, rowsum);
  norm_kernel<<<g, 256, 0, stream>>>(out, rowsum);
}
